// BigramLanguageModel_21045339750849
// MI455X (gfx1250) — matrix operand order repaired and checked by reading, not compiled, not run on hardware
//
#include <hip/hip_runtime.h>
#include <hip/hip_bf16.h>
#include <math.h>

typedef _Float16 half_t;
typedef __attribute__((ext_vector_type(16))) _Float16 v16h;
typedef __attribute__((ext_vector_type(8)))  float    v8f;

#define NB 32
#define NT 512
#define NE 384
#define NH 6
#define ND 64
#define NV 65
#define NL 6
#define NF 1536
#define NM (NB * NT)          // 16384 token rows
#define LN_EPS 1e-5f

#define WMMA_F16(A_, B_, C_) \
    __builtin_amdgcn_wmma_f32_16x16x32_f16(false, (A_), false, (B_), (short)0, (C_), \
                                           false, false)

// ---------------------------------------------------------------------------
// Embedding: x[b,t,:] = tok_emb[idx[b,t],:] + pos_emb[t,:]
// ---------------------------------------------------------------------------
__global__ void embed_kernel(const int* __restrict__ idx,
                             const float* __restrict__ tok,
                             const float* __restrict__ pos,
                             float* __restrict__ x) {
    int row = blockIdx.x;            // b*T + t
    int t   = row % NT;
    int tk  = idx[row];
    for (int e = threadIdx.x; e < NE; e += blockDim.x)
        x[(long)row * NE + e] = tok[(long)tk * NE + e] + pos[(long)t * NE + e];
}

// ---------------------------------------------------------------------------
// LayerNorm: one 128-thread block per row; f32 in -> f16 out (WMMA A operand)
// ---------------------------------------------------------------------------
__global__ __launch_bounds__(128)
void layernorm_kernel(const float* __restrict__ x,
                      const float* __restrict__ g,
                      const float* __restrict__ bta,
                      half_t* __restrict__ out) {
    __shared__ float red[128];
    int row = blockIdx.x;
    const float* xr = x + (long)row * NE;

    float s = 0.f;
    for (int e = threadIdx.x; e < NE; e += 128) s += xr[e];
    red[threadIdx.x] = s; __syncthreads();
    for (int off = 64; off > 0; off >>= 1) {
        if (threadIdx.x < off) red[threadIdx.x] += red[threadIdx.x + off];
        __syncthreads();
    }
    float mean = red[0] / NE;
    __syncthreads();

    float vs = 0.f;
    for (int e = threadIdx.x; e < NE; e += 128) { float d = xr[e] - mean; vs += d * d; }
    red[threadIdx.x] = vs; __syncthreads();
    for (int off = 64; off > 0; off >>= 1) {
        if (threadIdx.x < off) red[threadIdx.x] += red[threadIdx.x + off];
        __syncthreads();
    }
    float rstd = rsqrtf(red[0] / NE + LN_EPS);

    half_t* o = out + (long)row * NE;
    for (int e = threadIdx.x; e < NE; e += 128)
        o[e] = (half_t)((xr[e] - mean) * rstd * g[e] + bta[e]);
}

// ---------------------------------------------------------------------------
// Weight prep: f32 logical [K,N] (possibly head-blocked with block width
// bcols) -> f16 TRANSPOSED  out[n*K + k].  Makes every WMMA B-fragment a
// single contiguous 32-byte run per lane.
// ---------------------------------------------------------------------------
__global__ void convert_transpose_kernel(const float* __restrict__ in,
                                         half_t* __restrict__ out,
                                         int K, int N, int bcols) {
    long i = (long)blockIdx.x * blockDim.x + threadIdx.x;   // over N*K
    if (i >= (long)N * K) return;
    int n = (int)(i / K);
    int k = (int)(i % K);
    out[i] = (half_t)in[(long)(n / bcols) * K * bcols + (long)k * bcols + (n % bcols)];
}

// ---------------------------------------------------------------------------
// WMMA GEMM: C[M,N] = A[M,K] * Bt^T (+bias)(+residual)(+relu)
//   A : f16 row-major [M,lda];  Bt: f16 row-major [N,K] (transposed weights)
//   One wave computes a 16x64 tile.  Software-pipelined K loop: fragments for
//   step i+1 are loaded (fresh registers) BEFORE the 4 WMMAs of step i, so
//   waits cover loads that already have a full iteration of latency hiding.
//   omode 0: out[m*N+n]   omode 1: V-scatter out[((b*H+h)*D+d)*T+t]
// ---------------------------------------------------------------------------
__global__ __launch_bounds__(32)
void wmma_gemm_kernel(const half_t* __restrict__ A, int lda,
                      const half_t* __restrict__ Bt,
                      const float* __restrict__ bias,
                      const float* __restrict__ resid,
                      float* __restrict__ outF, half_t* __restrict__ outH,
                      int M, int N, int K, int relu, int omode) {
    int lane = threadIdx.x;
    int half_id = lane >> 4, lr = lane & 15;
    int m0 = blockIdx.y * 16;
    int n0 = blockIdx.x * 64;
    long arow = (long)(m0 + lr) * lda;

    // Per-subtile clamped B row pointers (loop-invariant, branch-free loads)
    const half_t* bp0;
    const half_t* bp1;
    const half_t* bp2;
    const half_t* bp3;
    {
        int n_0 = n0 + lr,      c0 = n_0 < N ? n_0 : N - 1;
        int n_1 = n0 + 16 + lr, c1 = n_1 < N ? n_1 : N - 1;
        int n_2 = n0 + 32 + lr, c2 = n_2 < N ? n_2 : N - 1;
        int n_3 = n0 + 48 + lr, c3 = n_3 < N ? n_3 : N - 1;
        long koff = 16 * half_id;
        bp0 = Bt + (long)c0 * K + koff;
        bp1 = Bt + (long)c1 * K + koff;
        bp2 = Bt + (long)c2 * K + koff;
        bp3 = Bt + (long)c3 * K + koff;
    }

    if (resid)   // warm the epilogue row (lowers to global_prefetch_b8)
        __builtin_prefetch(resid + (long)(m0 + lr) * N + n0, 0, 1);

    auto loadA = [&](v16h& a, int k0) {
        const half_t* al = A + arow + k0;
#pragma unroll
        for (int j = 0; j < 8; ++j) {
            a[j]     = al[half_id * 16 + j];               // k = 16h + j
            a[j + 8] = al[half_id * 16 + 8 + j];           // k = 16h + 8 + j
        }
    };
    auto loadB4 = [&](v16h& b0, v16h& b1, v16h& b2, v16h& b3, int k0) {
#pragma unroll
        for (int j = 0; j < 16; ++j) {                     // 4 x 32B contiguous
            b0[j] = bp0[k0 + j];
            b1[j] = bp1[k0 + j];
            b2[j] = bp2[k0 + j];
            b3[j] = bp3[k0 + j];
        }
    };

    v16h a, b0, b1, b2, b3;
    loadA(a, 0);
    loadB4(b0, b1, b2, b3, 0);

    v8f acc0 = {}, acc1 = {}, acc2 = {}, acc3 = {};
#pragma unroll 2
    for (int k0 = 0; k0 < K; k0 += 32) {
        int kn = (k0 + 32 < K) ? k0 + 32 : k0;   // last trip: re-load (discarded)
        v16h an, c0, c1, c2, c3;
        loadA(an, kn);
        loadB4(c0, c1, c2, c3, kn);
        acc0 = WMMA_F16(a, b0, acc0);
        acc1 = WMMA_F16(a, b1, acc1);
        acc2 = WMMA_F16(a, b2, acc2);
        acc3 = WMMA_F16(a, b3, acc3);
        a = an; b0 = c0; b1 = c1; b2 = c2; b3 = c3;
    }

    v8f accs[4] = {acc0, acc1, acc2, acc3};
#pragma unroll
    for (int s = 0; s < 4; ++s) {
        int nn = n0 + s * 16 + lr;
        if (nn < N) {
            float bv = bias ? bias[nn] : 0.f;
#pragma unroll
            for (int r = 0; r < 8; ++r) {
                int m = m0 + r + 8 * half_id;
                float vv = accs[s][r] + bv;
                if (resid) vv += resid[(long)m * N + nn];
                if (relu)  vv = fmaxf(vv, 0.f);
                long oi;
                if (omode == 1) {   // [B,H,D,T] scatter for V
                    int bb = m / NT, t = m % NT, h = nn / ND, d = nn % ND;
                    oi = (((long)bb * NH + h) * ND + d) * NT + t;
                } else {
                    oi = (long)m * N + nn;
                }
                if (outF) outF[oi] = vv;
                if (outH) outH[oi] = (half_t)vv;
            }
        }
    }
}

// ---------------------------------------------------------------------------
// Fused causal attention. One wave per (b, h, 16-query tile).
//   q,k,o: f16 [B,T,E] head-interleaved (col = h*64+d)
//   vt   : f16 [B,H,D,T]  (so P*V B-fragments load contiguously)
// ---------------------------------------------------------------------------
__global__ __launch_bounds__(32)
void attention_kernel(const half_t* __restrict__ q, const half_t* __restrict__ k,
                      const half_t* __restrict__ vt, half_t* __restrict__ o,
                      float scale) {
    __shared__ float  sbuf[16][NT];   // raw scores / exp
    __shared__ half_t pbuf[16][NT];   // normalized probs (WMMA A operand)

    int qt = blockIdx.x, h = blockIdx.y, b = blockIdx.z;
    int lane = threadIdx.x, half_id = lane >> 4, lr = lane & 15;
    int t0 = qt * 16;
    long rowbase = (long)b * NT;
    long hoff    = (long)h * ND;
    long vtbase  = ((long)b * NH + h) * (long)ND * NT;

    // Q fragments: 16x64 = two 16x32 A-fragments (contiguous 16-half runs)
    v16h aq0, aq1;
    {
        long qrow = (rowbase + t0 + lr) * NE + hoff;
#pragma unroll
        for (int j = 0; j < 8; ++j) {
            aq0[j]     = q[qrow + half_id * 16 + j];
            aq0[j + 8] = q[qrow + half_id * 16 + 8 + j];
            aq1[j]     = q[qrow + 32 + half_id * 16 + j];
            aq1[j + 8] = q[qrow + 32 + half_id * 16 + 8 + j];
        }
    }

    // ---- scores over all causal key tiles ----
    for (int kt = 0; kt <= qt; ++kt) {
        long krow = (rowbase + kt * 16 + lr) * NE + hoff;  // B col n = key
        v16h b0, b1;
#pragma unroll
        for (int j = 0; j < 16; ++j) {
            b0[j] = k[krow + 16 * half_id + j];            // d = WMMA K dim
            b1[j] = k[krow + 32 + 16 * half_id + j];
        }
        v8f c = {};
        c = WMMA_F16(aq0, b0, c);
        c = WMMA_F16(aq1, b1, c);
        int key = kt * 16 + lr;
#pragma unroll
        for (int r = 0; r < 8; ++r) {
            int m = r + 8 * half_id;
            sbuf[m][key] = (key <= t0 + m) ? c[r] * scale : -__builtin_inff();
        }
    }
    __syncthreads();

    int Lq   = (qt + 1) * 16;            // valid keys
    int Lpad = (Lq + 31) & ~31;          // pad K-dim to WMMA multiple

    // ---- masked softmax, row by row, 32 lanes cooperating ----
    for (int m = 0; m < 16; ++m) {
        float mx = -__builtin_inff();
        for (int cix = lane; cix < Lq; cix += 32) mx = fmaxf(mx, sbuf[m][cix]);
        for (int off = 16; off > 0; off >>= 1) mx = fmaxf(mx, __shfl_xor(mx, off, 32));
        float sum = 0.f;
        for (int cix = lane; cix < Lq; cix += 32) {
            float e = __expf(sbuf[m][cix] - mx);
            sum += e;
            sbuf[m][cix] = e;
        }
        for (int off = 16; off > 0; off >>= 1) sum += __shfl_xor(sum, off, 32);
        float inv = 1.f / sum;
        for (int cix = lane; cix < Lq; cix += 32)
            pbuf[m][cix] = (half_t)(sbuf[m][cix] * inv);
        for (int cix = Lq + lane; cix < Lpad; cix += 32)
            pbuf[m][cix] = (half_t)0.f;  // padded keys contribute nothing
    }
    __syncthreads();

    // ---- O = P V : accumulate 16x64 in four 16x16 column tiles ----
    v8f co0 = {}, co1 = {}, co2 = {}, co3 = {};
    const half_t* vcol = vt + vtbase + (long)lr * NT + 16 * half_id;
    for (int kc = 0; kc < Lpad; kc += 32) {
        v16h ap, bv0, bv1, bv2, bv3;
#pragma unroll
        for (int j = 0; j < 8; ++j) {
            ap[j]     = pbuf[lr][kc + half_id * 16 + j];
            ap[j + 8] = pbuf[lr][kc + half_id * 16 + 8 + j];
        }
#pragma unroll
        for (int j = 0; j < 16; ++j) {                     // contiguous key runs
            bv0[j] = vcol[kc + j];
            bv1[j] = vcol[(long)16 * NT + kc + j];
            bv2[j] = vcol[(long)32 * NT + kc + j];
            bv3[j] = vcol[(long)48 * NT + kc + j];
        }
        co0 = WMMA_F16(ap, bv0, co0);
        co1 = WMMA_F16(ap, bv1, co1);
        co2 = WMMA_F16(ap, bv2, co2);
        co3 = WMMA_F16(ap, bv3, co3);
    }
    v8f cos[4] = {co0, co1, co2, co3};
#pragma unroll
    for (int dt = 0; dt < 4; ++dt)
#pragma unroll
        for (int r = 0; r < 8; ++r) {
            int t = t0 + r + 8 * half_id;
            o[(rowbase + t) * NE + hoff + dt * 16 + lr] = (half_t)cos[dt][r];
        }
}

// ---------------------------------------------------------------------------
// Host orchestration
// ---------------------------------------------------------------------------
extern "C" void kernel_launch(void* const* d_in, const int* in_sizes, int n_in,
                              void* d_out, int out_size, void* d_ws, size_t ws_size,
                              hipStream_t stream) {
    (void)in_sizes; (void)n_in; (void)out_size; (void)ws_size;

    const int*   idx   = (const int*)d_in[0];
    const float* tok   = (const float*)d_in[1];
    const float* pos   = (const float*)d_in[2];
    const float* Wq    = (const float*)d_in[3];
    const float* Wk    = (const float*)d_in[4];
    const float* Wv    = (const float*)d_in[5];
    const float* Wproj = (const float*)d_in[6];
    const float* bproj = (const float*)d_in[7];
    const float* W1    = (const float*)d_in[8];
    const float* b1    = (const float*)d_in[9];
    const float* W2    = (const float*)d_in[10];
    const float* b2    = (const float*)d_in[11];
    const float* ln1g  = (const float*)d_in[12];
    const float* ln1b  = (const float*)d_in[13];
    const float* ln2g  = (const float*)d_in[14];
    const float* ln2b  = (const float*)d_in[15];
    const float* lnfg  = (const float*)d_in[16];
    const float* lnfb  = (const float*)d_in[17];
    const float* Wlm   = (const float*)d_in[18];
    const float* blm   = (const float*)d_in[19];
    float* out = (float*)d_out;

    // Workspace carve-up (256B aligned)
    char* w = (char*)d_ws;
    auto carve = [&](size_t bytes) {
        char* p = w;
        w += (bytes + 255) & ~(size_t)255;
        return p;
    };
    float*  x    = (float*) carve((size_t)NM * NE * 4);
    half_t* hbuf = (half_t*)carve((size_t)NM * NE * 2);
    half_t* qb   = (half_t*)carve((size_t)NM * NE * 2);
    half_t* kbb  = (half_t*)carve((size_t)NM * NE * 2);
    half_t* vtb  = (half_t*)carve((size_t)NM * NE * 2);   // [B,H,D,T]
    half_t* ab   = (half_t*)carve((size_t)NM * NE * 2);
    half_t* hid  = (half_t*)carve((size_t)NM * NF * 2);
    half_t* wq16 = (half_t*)carve((size_t)NL * NE * NE * 2);   // [N=E][K=E] per layer
    half_t* wk16 = (half_t*)carve((size_t)NL * NE * NE * 2);
    half_t* wv16 = (half_t*)carve((size_t)NL * NE * NE * 2);
    half_t* wp16 = (half_t*)carve((size_t)NL * NE * NE * 2);
    half_t* w116 = (half_t*)carve((size_t)NL * NE * NF * 2);   // [NF][NE]
    half_t* w216 = (half_t*)carve((size_t)NL * NF * NE * 2);   // [NE][NF]
    half_t* wl16 = (half_t*)carve((size_t)NV * NE * 2);        // [NV][NE]

    auto convT = [&](const float* src, half_t* dst, int K, int N, int bcols) {
        long n = (long)N * K;
        convert_transpose_kernel<<<dim3((unsigned)((n + 255) / 256)), dim3(256), 0,
                                   stream>>>(src, dst, K, N, bcols);
    };
    for (int l = 0; l < NL; ++l) {
        size_t wo = (size_t)l * NE * NE;
        convT(Wq + (size_t)l * NH * NE * ND, wq16 + wo, NE, NE, ND);
        convT(Wk + (size_t)l * NH * NE * ND, wk16 + wo, NE, NE, ND);
        convT(Wv + (size_t)l * NH * NE * ND, wv16 + wo, NE, NE, ND);
        convT(Wproj + (size_t)l * NE * NE,   wp16 + wo, NE, NE, NE);
        convT(W1 + (size_t)l * NE * NF, w116 + (size_t)l * NE * NF, NE, NF, NF);
        convT(W2 + (size_t)l * NF * NE, w216 + (size_t)l * NF * NE, NF, NE, NE);
    }
    convT(Wlm, wl16, NE, NV, NV);

    auto gemm = [&](const half_t* A, int lda, const half_t* Bt,
                    const float* bias, const float* resid, float* oF, half_t* oH,
                    int M, int N, int K, int relu, int omode) {
        dim3 g((N + 63) / 64, M / 16);
        wmma_gemm_kernel<<<g, dim3(32), 0, stream>>>(A, lda, Bt, bias, resid,
                                                     oF, oH, M, N, K, relu, omode);
    };

    embed_kernel<<<dim3(NM), dim3(128), 0, stream>>>(idx, tok, pos, x);

    float scale = 1.f / sqrtf((float)NE);   // reference scales by n_embed^-0.5
    for (int l = 0; l < NL; ++l) {
        size_t wo = (size_t)l * NE * NE;
        layernorm_kernel<<<dim3(NM), dim3(128), 0, stream>>>(x, ln1g + l * NE,
                                                             ln1b + l * NE, hbuf);
        gemm(hbuf, NE, wq16 + wo, nullptr, nullptr, nullptr, qb,  NM, NE, NE, 0, 0);
        gemm(hbuf, NE, wk16 + wo, nullptr, nullptr, nullptr, kbb, NM, NE, NE, 0, 0);
        gemm(hbuf, NE, wv16 + wo, nullptr, nullptr, nullptr, vtb, NM, NE, NE, 0, 1);

        attention_kernel<<<dim3(NT / 16, NH, NB), dim3(32), 0, stream>>>(qb, kbb, vtb,
                                                                         ab, scale);
        // out-proj + bias + residual (in place on x: each element RMW by one lane)
        gemm(ab, NE, wp16 + wo, bproj + l * NE, x, x, nullptr, NM, NE, NE, 0, 0);

        layernorm_kernel<<<dim3(NM), dim3(128), 0, stream>>>(x, ln2g + l * NE,
                                                             ln2b + l * NE, hbuf);
        // MLP up + ReLU
        gemm(hbuf, NE, w116 + (size_t)l * NE * NF, b1 + l * NF, nullptr, nullptr,
             hid, NM, NF, NE, 1, 0);
        // MLP down + bias + residual
        gemm(hid, NF, w216 + (size_t)l * NF * NE, b2 + l * NE, x, x, nullptr,
             NM, NE, NF, 0, 0);
    }

    layernorm_kernel<<<dim3(NM), dim3(128), 0, stream>>>(x, lnfg, lnfb, hbuf);
    // LM head -> f32 logits [B,T,65]
    gemm(hbuf, NE, wl16, blm, nullptr, out, nullptr, NM, NV, NE, 0, 0);
}
